// Custom_GATConv_v2_62268435857940
// MI455X (gfx1250) — matrix operand order repaired and checked by reading, not compiled, not run on hardware
//
#include <hip/hip_runtime.h>
#include <hip/hip_bf16.h>
#include <stdint.h>
#include <stddef.h>

// Problem constants (reference: B=2, N=4096, C_IN=256, HEADS=4, D_OUT=64)
#define B_    2
#define N_    4096
#define CIN   256
#define HEADS 4
#define DOUT  64
#define COUT  256            // HEADS*DOUT
#define SLOPE 0.2f

typedef __attribute__((ext_vector_type(16))) _Float16     v16h;
typedef __attribute__((ext_vector_type(8)))  float        v8f;
typedef __attribute__((ext_vector_type(4)))  unsigned int u32x4;
typedef __attribute__((ext_vector_type(2)))  unsigned int u32x2;
typedef __attribute__((ext_vector_type(4)))  float        f32x4;

union HFrag  { v16h v; u32x4 u[2]; _Float16 h[16]; };
union HPack8 { _Float16 h[8]; u32x4 u; };

// ---------------------------------------------------------------- converters
__global__ void k_cvt_x(const float* __restrict__ x, _Float16* __restrict__ xh, int n) {
  int t = blockIdx.x * 256 + threadIdx.x;
  if (t < n) xh[t] = (_Float16)x[t];
}

// W is (K=256, C=256) row-major; produce W_t (C, K) in f16 so B-fragments are contiguous in K.
__global__ void k_cvt_w(const float* __restrict__ W, _Float16* __restrict__ wt) {
  int t = blockIdx.x * 256 + threadIdx.x;   // t = c*256 + k
  int c = t >> 8, k = t & 255;
  wt[t] = (_Float16)W[k * COUT + c];
}

// ---------------------------------------------------------------- GEMM: Wh = x @ W  (WMMA f16)
// Each wave computes a 16(token) x 64(channel) strip; output stored TRANSPOSED:
// wh_t[((b*H + h)*64 + d) * N + tok]  (f16), so the PV kernel's B-fragments are b128 loads.
__global__ void __launch_bounds__(256) k_gemm(const _Float16* __restrict__ xh,
                                              const _Float16* __restrict__ wt,
                                              _Float16* __restrict__ wh_t) {
  const int lane = threadIdx.x & 31;
  const int laneHi = lane >> 4;
  const int r = lane & 15;
  const int gw = blockIdx.x * 8 + (threadIdx.x >> 5);   // 2048 waves
  const int rowTile = gw >> 2;                          // 0..511  (B*N/16)
  const int colGroup = gw & 3;                          // 0..3    (256/64)
  const int i0 = rowTile * 16;
  const int c0 = colGroup * 64;

  v8f acc[4] = {};
  const _Float16* arow = xh + (size_t)(i0 + r) * CIN + laneHi * 8;

  for (int k0 = 0; k0 < CIN; k0 += 32) {
    HFrag A;
    const u32x4* ap = (const u32x4*)(arow + k0);
    A.u[0] = ap[0];        // K = k0 + laneHi*8 + 0..7   -> VGPR 0..3
    A.u[1] = ap[2];        // K = k0 + 16 + laneHi*8 + 0..7 -> VGPR 4..7
#pragma unroll
    for (int ct = 0; ct < 4; ++ct) {
      int c = c0 + ct * 16 + r;
      const u32x4* bp = (const u32x4*)(wt + (size_t)c * CIN + k0 + laneHi * 8);
      HFrag Bf;
      Bf.u[0] = bp[0];     // K = k0 + laneHi*8 + 0..7
      Bf.u[1] = bp[2];     // K = k0 + 16 + laneHi*8 + 0..7
      acc[ct] = __builtin_amdgcn_wmma_f32_16x16x32_f16(false, A.v, false, Bf.v,
                                                       (short)0, acc[ct], false, false);
    }
  }

  const int b = i0 >> 12;                    // i0 / 4096
  const int tok0 = (i0 & (N_ - 1)) + laneHi * 8;
#pragma unroll
  for (int ct = 0; ct < 4; ++ct) {
    int c = c0 + ct * 16 + r;
    int bh = b * HEADS + (c >> 6);
    int d = c & 63;
    HPack8 pk;
#pragma unroll
    for (int v = 0; v < 8; ++v) pk.h[v] = (_Float16)acc[ct][v];   // rows m=v+8*laneHi -> consecutive tokens
    *(u32x4*)(wh_t + ((size_t)bh * DOUT + d) * N_ + tok0) = pk.u;
  }
}

// ---------------------------------------------------------------- src/tgt = Wh . a_{src,tgt}
__global__ void k_src_tgt(const _Float16* __restrict__ wh_t, const float* __restrict__ att,
                          float* __restrict__ src, float* __restrict__ tgt) {
  int t = blockIdx.x * 256 + threadIdx.x;   // t = bh*4096 + i
  int bh = t >> 12;
  int h = bh & (HEADS - 1);
  int i = t & (N_ - 1);
  const float* as = att + h * 2 * DOUT;
  const float* at = as + DOUT;
  const _Float16* col = wh_t + (size_t)bh * DOUT * N_ + i;
  float s = 0.f, g = 0.f;
#pragma unroll 4
  for (int d = 0; d < DOUT; ++d) {
    float w = (float)col[(size_t)d * N_];
    s += w * as[d];
    g += w * at[d];
  }
  src[t] = s;
  tgt[t] = g;
}

// ---------------------------------------------------------------- per-row softmax stats (m, l)
// One wave per (b,h,i) row; online max/sum + 5-step shuffle reduction.
__global__ void __launch_bounds__(256) k_stats(const float* __restrict__ src,
                                               const float* __restrict__ tgt,
                                               const unsigned char* __restrict__ adj,
                                               float* __restrict__ mbuf, float* __restrict__ lbuf) {
  const int lane = threadIdx.x & 31;
  const int rid = blockIdx.x * 8 + (threadIdx.x >> 5);   // 0..32767
  const int bh = rid >> 12;
  const int b = bh >> 2;
  const int i = rid & (N_ - 1);
  const float s = src[rid];
  const float* trow = tgt + (size_t)bh * N_;
  const unsigned char* arow = adj + ((size_t)b * N_ + i) * N_;

  float mx = -1e30f, ls = 0.f;
  for (int it = 0; it < N_ / 32; ++it) {
    int j = lane + it * 32;
    float xv = s + trow[j];
    float e = fmaxf(xv, SLOPE * xv);          // leaky_relu, slope 0.2
    if (!arow[j]) {                           // adj True => masked out (-inf)
      float nm = fmaxf(mx, e);
      ls = ls * __expf(mx - nm) + __expf(e - nm);
      mx = nm;
    }
  }
#pragma unroll
  for (int off = 16; off; off >>= 1) {
    float om = __shfl_xor(mx, off);
    float ol = __shfl_xor(ls, off);
    float nm = fmaxf(mx, om);
    ls = ls * __expf(mx - nm) + ol * __expf(om - nm);
    mx = nm;
  }
  if (lane == 0) { mbuf[rid] = mx; lbuf[rid] = ls; }
}

// ---------------------------------------------------------------- fused softmax-PV (WMMA)
// One block per (b, 16-row i-tile). 8 waves split the j-chunks; heads accumulated in-register.
// P is generated directly in WMMA A-fragment layout (each A-lane owns one row i, so the
// precomputed m_i / l_i make normalization lane-local: no transpose, no cross-lane ops).
__global__ void __launch_bounds__(256) k_pv(const _Float16* __restrict__ wh_t,
                                            const float* __restrict__ src,
                                            const float* __restrict__ tgt,
                                            const float* __restrict__ mbuf,
                                            const float* __restrict__ lbuf,
                                            const unsigned char* __restrict__ adj,
                                            float* __restrict__ out) {
  __shared__ float s_tgt[N_];           // 16 KB
  __shared__ float s_red[8][16 * 64];   // 32 KB deterministic cross-wave reduction

  const int tid = threadIdx.x;
  const int lane = tid & 31;
  const int w = tid >> 5;
  const int laneHi = lane >> 4;
  const int r = lane & 15;
  const int b = blockIdx.x >> 8;               // 512 blocks = B * N/16
  const int i0 = (blockIdx.x & 255) << 4;
  const int i = i0 + r;                        // this lane's row
  const unsigned char* arow = adj + ((size_t)b * N_ + i) * N_;

  v8f acc[4] = {};
  for (int h = 0; h < HEADS; ++h) {
    const int bh = b * HEADS + h;
    for (int idx = tid; idx < N_; idx += 256) s_tgt[idx] = tgt[(size_t)bh * N_ + idx];
    __syncthreads();

    const float s = src[bh * N_ + i];
    const float mv = mbuf[bh * N_ + i];
    const float il = 1.f / lbuf[bh * N_ + i];
    const _Float16* vbase = wh_t + (size_t)bh * DOUT * N_;

    for (int c = w; c < N_ / 32; c += 8) {     // 16 chunks per wave, uniform
      const int jc = c * 32;
      HFrag A;
#pragma unroll
      for (int run = 0; run < 2; ++run) {
        int jg = jc + run * 16 + laneHi * 8;   // 8 consecutive j for this lane
        f32x4 ta = *(const f32x4*)&s_tgt[jg];
        f32x4 tb = *(const f32x4*)&s_tgt[jg + 4];
        u32x2 mk = *(const u32x2*)(arow + jg); // 8 mask bytes
#pragma unroll
        for (int t8 = 0; t8 < 8; ++t8) {
          float tv = (t8 < 4) ? ta[t8] : tb[t8 - 4];
          unsigned int mb = (mk[t8 >> 2] >> ((t8 & 3) * 8)) & 0xFFu;
          float xv = s + tv;
          float e = fmaxf(xv, SLOPE * xv);
          float p = mb ? 0.f : __expf(e - mv) * il;
          A.h[(run * 4 + (t8 >> 1)) * 2 + (t8 & 1)] = (_Float16)p;
        }
      }
#pragma unroll
      for (int dt = 0; dt < 4; ++dt) {
        const u32x4* bp = (const u32x4*)(vbase + ((size_t)(dt * 16 + r)) * N_ + jc + laneHi * 8);
        HFrag Bf;
        Bf.u[0] = bp[0];
        Bf.u[1] = bp[2];
        acc[dt] = __builtin_amdgcn_wmma_f32_16x16x32_f16(false, A.v, false, Bf.v,
                                                         (short)0, acc[dt], false, false);
      }
    }
    __syncthreads();
  }

  // per-wave partials -> LDS, then fixed-order sum (deterministic), mean over heads
#pragma unroll
  for (int dt = 0; dt < 4; ++dt)
#pragma unroll
    for (int v = 0; v < 8; ++v)
      s_red[w][(v + 8 * laneHi) * 64 + dt * 16 + r] = acc[dt][v];
  __syncthreads();

  for (int el = tid; el < 16 * 64; el += 256) {
    float sum = 0.f;
#pragma unroll
    for (int ww = 0; ww < 8; ++ww) sum += s_red[ww][el];
    int row = el >> 6, d = el & 63;
    out[((size_t)b * N_ + i0 + row) * DOUT + d] = 0.25f * sum;
  }
}

// ---------------------------------------------------------------- launcher
extern "C" void kernel_launch(void* const* d_in, const int* in_sizes, int n_in,
                              void* d_out, int out_size, void* d_ws, size_t ws_size,
                              hipStream_t stream) {
  const float* x = (const float*)d_in[0];
  const unsigned char* adj = (const unsigned char*)d_in[1];  // numpy bool: 1 byte/elem
  const float* W = (const float*)d_in[2];
  const float* att = (const float*)d_in[3];
  float* out = (float*)d_out;

  char* ws = (char*)d_ws;
  size_t off = 0;
  _Float16* xh = (_Float16*)(ws + off);  off += (size_t)B_ * N_ * CIN * 2;       // 4 MB
  _Float16* wt = (_Float16*)(ws + off);  off += (size_t)CIN * COUT * 2;          // 128 KB
  _Float16* wh = (_Float16*)(ws + off);  off += (size_t)B_ * HEADS * DOUT * N_ * 2; // 4 MB
  float* srcb = (float*)(ws + off);      off += (size_t)B_ * HEADS * N_ * 4;     // 128 KB
  float* tgtb = (float*)(ws + off);      off += (size_t)B_ * HEADS * N_ * 4;
  float* mb   = (float*)(ws + off);      off += (size_t)B_ * HEADS * N_ * 4;
  float* lb   = (float*)(ws + off);      off += (size_t)B_ * HEADS * N_ * 4;
  (void)ws_size; (void)in_sizes; (void)n_in; (void)out_size;

  const int nx = B_ * N_ * CIN;
  k_cvt_x<<<nx / 256, 256, 0, stream>>>(x, xh, nx);
  k_cvt_w<<<(CIN * COUT) / 256, 256, 0, stream>>>(W, wt);
  k_gemm<<<256, 256, 0, stream>>>(xh, wt, wh);                       // 2048 waves
  k_src_tgt<<<(B_ * HEADS * N_) / 256, 256, 0, stream>>>(wh, att, srcb, tgtb);
  k_stats<<<(B_ * HEADS * N_) / 8, 256, 0, stream>>>(srcb, tgtb, adj, mb, lb);
  k_pv<<<B_ * (N_ / 16), 256, 0, stream>>>(wh, srcb, tgtb, mb, lb, adj, out);
}
